// WindowAttention_83339545412148
// MI455X (gfx1250) — compile-verified, matrix operand order checked by reading, not run on hardware
//
#include <hip/hip_runtime.h>
#include <stdint.h>

// ---------------- problem constants ----------------
#define WIN_N   49          // tokens per window
#define NP      64          // padded tokens (4 x 16 WMMA tiles)
#define CDIM    384
#define NHEADS  12
#define HDIM    32
#define QKV_COLS 1152
#define ATTN_SCALE 0.17677669529663687f  // 32^-0.5

// LDS strides (elements). All chosen so (row * LD * 2B) is a multiple of 16B.
#define LDA 392   // sA / sO row stride (384 + 8)   -> 784 B/row
#define LDK 40    // sWt row stride (32 K + 8)      -> 80 B/row
#define LDQ 40    // sQ/sK row stride (32 + 8)      -> 80 B/row
#define LDV 72    // sVt row stride (64 tok + 8)    -> 144 B/row
#define LDS_S 68  // sS (f32) row stride (64 + 4)
#define LDP 72    // sP row stride (64 + 8)         -> 144 B/row
#define WPANEL (96 * LDK)   // one weight-panel buffer (3840 elements)

typedef __attribute__((ext_vector_type(16))) __bf16 bf16x16;
typedef __attribute__((ext_vector_type(8)))  __bf16 bf16x8;
typedef __attribute__((ext_vector_type(8)))  float  floatx8;
typedef __attribute__((ext_vector_type(4)))  float  floatx4;

__device__ __forceinline__ __bf16 f2bf(float f) { return (__bf16)f; }  // v_cvt RNE
__device__ __forceinline__ unsigned short f2bf_u(float f) {
  return __builtin_bit_cast(unsigned short, (__bf16)f);
}
__device__ __forceinline__ floatx8 fzero() {
  floatx8 z;
#pragma unroll
  for (int e = 0; e < 8; ++e) z[e] = 0.0f;
  return z;
}
__device__ __forceinline__ floatx8 wmma_bf16(bf16x16 a, bf16x16 b, floatx8 c) {
  // (neg_a, A, neg_b, B, c_mod, C, reuse_a, reuse_b)
  return __builtin_amdgcn_wmma_f32_16x16x32_bf16(false, a, false, b, (short)0, c,
                                                 false, false);
}

// Fragment loader for a row-major MxK source (A matrix), also used for
// B = M^T (lane then indexes the N column = source row). Per CDNA5 ISA:
// lane<16 -> row=lane,    K in {k0..k0+7, k0+16..k0+23}
// lane>=16-> row=lane-16, K in {k0+8..k0+15, k0+24..k0+31}
// Both 8-element groups are contiguous & 16B aligned -> two ds_load_b128.
template <int LD>
__device__ __forceinline__ bf16x16 frag_rowK(const unsigned short* M, int row0,
                                             int k0, int lane) {
  const unsigned short* p = M + (row0 + (lane & 15)) * LD + k0 + ((lane >> 4) << 3);
  bf16x8 lo = *reinterpret_cast<const bf16x8*>(p);
  bf16x8 hi = *reinterpret_cast<const bf16x8*>(p + 16);
  return __builtin_shufflevector(lo, hi, 0, 1, 2, 3, 4, 5, 6, 7, 8, 9, 10, 11, 12,
                                 13, 14, 15);
}

// Async global->LDS 16-byte copy (gfx1250 GLOBAL_LOAD_ASYNC_TO_LDS_B128,
// tracked on ASYNCcnt). lds_off = byte offset within the group segment.
__device__ __forceinline__ void async_copy_b128(unsigned lds_off, const void* gsrc) {
  asm volatile("global_load_async_to_lds_b128 %0, %1, off"
               :
               : "v"(lds_off), "v"(gsrc)
               : "memory");
}
__device__ __forceinline__ void async_wait0() {
  asm volatile("s_wait_asynccnt 0" ::: "memory");
}

// ---------------- pre-pass: convert + transpose weights, gather bias ----------
__global__ __launch_bounds__(256) void convert_transpose_kernel(
    const float* __restrict__ qkv_w, const float* __restrict__ proj_w,
    unsigned short* __restrict__ wqT, unsigned short* __restrict__ wpT) {
  int i = blockIdx.x * 256 + threadIdx.x;
  if (i < QKV_COLS * CDIM) {            // wqT[col][k] = qkv_w[k][col]
    int col = i / CDIM, k = i % CDIM;
    wqT[i] = f2bf_u(qkv_w[(size_t)k * QKV_COLS + col]);
  }
  if (i < CDIM * CDIM) {                // wpT[col][k] = proj_w[k][col]
    int col = i / CDIM, k = i % CDIM;
    wpT[i] = f2bf_u(proj_w[(size_t)k * CDIM + col]);
  }
}

__global__ __launch_bounds__(256) void gather_bias_kernel(
    const float* __restrict__ rpb, const int* __restrict__ rel,
    float* __restrict__ bias_out) {
  int i = blockIdx.x * 256 + threadIdx.x;        // over NHEADS * 49 * 49
  if (i < NHEADS * WIN_N * WIN_N) {
    int h = i / (WIN_N * WIN_N);
    int ij = i % (WIN_N * WIN_N);
    bias_out[i] = rpb[rel[ij] * NHEADS + h];
  }
}

// ---------------- fused window-attention kernel ----------------
__global__ __launch_bounds__(256) void fused_window_attn_kernel(
    const float* __restrict__ x, const float* __restrict__ mask,
    const unsigned short* __restrict__ wqT, const float* __restrict__ qkv_b,
    const unsigned short* __restrict__ wpT, const float* __restrict__ proj_b,
    const float* __restrict__ bias_hij, float* __restrict__ out, int nW) {
  __shared__ __attribute__((aligned(16))) unsigned short sA[NP * LDA];    // x bf16
  __shared__ __attribute__((aligned(16))) unsigned short sWt[2 * WPANEL]; // W^T dbuf
  __shared__ __attribute__((aligned(16))) unsigned short sQ[NP * LDQ];    // Q scaled
  __shared__ __attribute__((aligned(16))) unsigned short sK[NP * LDQ];
  __shared__ __attribute__((aligned(16))) unsigned short sVt[HDIM * LDV]; // V^T
  __shared__ __attribute__((aligned(16))) float          sS[NP * LDS_S];  // scores
  __shared__ __attribute__((aligned(16))) unsigned short sP[NP * LDP];    // probs
  __shared__ __attribute__((aligned(16))) unsigned short sO[NP * LDA];    // attn out

  const int tid  = threadIdx.x;
  const int lane = tid & 31;
  const int wave = tid >> 5;
  const int b    = blockIdx.x;

  // ---- stage x window -> sA (bf16, rows >=49 zeroed); 8 elems per thread ----
  const float* xb = x + (size_t)b * WIN_N * CDIM;
  for (int idx = tid; idx < NP * CDIM / 8; idx += 256) {
    int r = idx / (CDIM / 8);
    int c = (idx % (CDIM / 8)) * 8;
    bf16x8 v8;
    if (r < WIN_N) {
      floatx4 f0 = *reinterpret_cast<const floatx4*>(xb + r * CDIM + c);
      floatx4 f1 = *reinterpret_cast<const floatx4*>(xb + r * CDIM + c + 4);
#pragma unroll
      for (int e = 0; e < 4; ++e) { v8[e] = f2bf(f0[e]); v8[4 + e] = f2bf(f1[e]); }
    } else {
#pragma unroll
      for (int e = 0; e < 8; ++e) v8[e] = f2bf(0.0f);
    }
    *reinterpret_cast<bf16x8*>(&sA[r * LDA + c]) = v8;
  }
  __syncthreads();

  // ---- per-head fused QKV + attention ----
  for (int h = 0; h < NHEADS; ++h) {
    // async-stage one W^T K-panel (96 rows = q|k|v cols of head h, 32 K) to buf
    auto stage_qkv_panel = [&](int buf, int ks) {
      unsigned short* dst = sWt + buf * WPANEL;
      for (int idx = tid; idx < 96 * 4; idx += 256) {
        int c = idx >> 2, q = idx & 3;                 // panel row, 16B chunk
        int matc = c >> 5, cc = c & 31;
        int gcol = matc * CDIM + h * HDIM + cc;
        const unsigned short* src = wqT + (size_t)gcol * CDIM + ks * 32 + q * 8;
        unsigned lds_off = (unsigned)(uintptr_t)(&dst[c * LDK + q * 8]);
        async_copy_b128(lds_off, src);
      }
    };

    // warm GL2 for the next head's weights
    if (h + 1 < NHEADS) {
      const unsigned short* nxt =
          wqT + (size_t)((h + 1) * HDIM + (tid % 96)) * CDIM;
      __builtin_prefetch((const void*)nxt, 0, 0);
    }

    // QKV GEMM: wave owns 16x16 tile (rt,ct) of Q, K and V head tiles.
    const int rt = wave & 3;        // row tile (4 x 16 rows)
    const int ct = wave >> 2;       // col tile (2 x 16 cols within Dh=32)
    floatx8 acc[3];
#pragma unroll
    for (int m = 0; m < 3; ++m) acc[m] = fzero();

    stage_qkv_panel(0, 0);          // prologue
    async_wait0();
    __syncthreads();

    for (int ks = 0; ks < CDIM / 32; ++ks) {
      const int cur = ks & 1;
      if (ks + 1 < CDIM / 32) stage_qkv_panel(cur ^ 1, ks + 1);  // overlap DMA

      const unsigned short* wbuf = sWt + cur * WPANEL;
      bf16x16 a = frag_rowK<LDA>(sA, rt * 16, ks * 32, lane);
#pragma unroll
      for (int m = 0; m < 3; ++m) {
        bf16x16 bw = frag_rowK<LDK>(wbuf, m * 32 + ct * 16, 0, lane);
        acc[m] = wmma_bf16(a, bw, acc[m]);
      }
      async_wait0();
      __syncthreads();
    }

    // write Q (pre-scaled) / K row-major, V transposed (contiguous 16B store)
    {
      const int col = ct * 16 + (lane & 15);
      const float bq = qkv_b[0 * CDIM + h * HDIM + col];
      const float bk = qkv_b[1 * CDIM + h * HDIM + col];
      const float bv = qkv_b[2 * CDIM + h * HDIM + col];
      const int rbase = rt * 16 + ((lane >> 4) << 3);
      bf16x8 vvec;
#pragma unroll
      for (int r = 0; r < 8; ++r) {
        int row = rbase + r;
        sQ[row * LDQ + col] = f2bf_u((acc[0][r] + bq) * ATTN_SCALE);
        sK[row * LDQ + col] = f2bf_u(acc[1][r] + bk);
        vvec[r] = f2bf(acc[2][r] + bv);
      }
      *reinterpret_cast<bf16x8*>(&sVt[col * LDV + rbase]) = vvec;  // V^T[d][tok]
    }
    __syncthreads();

    // scores S = Q * K^T : 16 tiles, 2 per wave, single K=32 WMMA each
#pragma unroll
    for (int j2 = 0; j2 < 2; ++j2) {
      int t = wave * 2 + j2;
      int srt = t >> 2, sct = t & 3;
      bf16x16 aq = frag_rowK<LDQ>(sQ, srt * 16, 0, lane);
      bf16x16 bk = frag_rowK<LDQ>(sK, sct * 16, 0, lane);  // B = K^T
      floatx8 s = wmma_bf16(aq, bk, fzero());
      const int col = sct * 16 + (lane & 15);
      const int rbase = srt * 16 + ((lane >> 4) << 3);
#pragma unroll
      for (int r = 0; r < 8; ++r) sS[(rbase + r) * LDS_S + col] = s[r];
    }
    __syncthreads();

    // softmax with bias + mask: 4 lanes per row (16 cols each),
    // row max/sum reduced with shfl_xor(1|2) inside the 4-lane group.
    {
      const int i   = tid >> 2;     // row 0..63
      const int sub = tid & 3;      // 16-column slice
      const int j0  = sub * 16;
      if (i < WIN_N) {
        const int jn = (j0 + 16 < WIN_N) ? j0 + 16 : WIN_N;
        const float* brow = bias_hij + ((size_t)h * WIN_N + i) * WIN_N;
        const float* mrow = mask + ((size_t)(b % nW) * WIN_N + i) * WIN_N;
        float mx = -1e30f;
        for (int j = j0; j < jn; ++j) {
          float s = sS[i * LDS_S + j] + brow[j] + mrow[j];
          sS[i * LDS_S + j] = s;
          mx = fmaxf(mx, s);
        }
        mx = fmaxf(mx, __shfl_xor(mx, 1, 32));
        mx = fmaxf(mx, __shfl_xor(mx, 2, 32));
        float sum = 0.0f;
        for (int j = j0; j < jn; ++j) {
          float e = __expf(sS[i * LDS_S + j] - mx);
          sS[i * LDS_S + j] = e;
          sum += e;
        }
        sum += __shfl_xor(sum, 1, 32);
        sum += __shfl_xor(sum, 2, 32);
        float inv = 1.0f / sum;
#pragma unroll
        for (int jj = 0; jj < 16; jj += 2) {   // packed bf16 pair stores
          int j = j0 + jj;
          float p0 = (j     < WIN_N) ? sS[i * LDS_S + j] * inv : 0.0f;
          float p1 = (j + 1 < WIN_N) ? sS[i * LDS_S + j + 1] * inv : 0.0f;
          unsigned pk = (unsigned)f2bf_u(p0) | ((unsigned)f2bf_u(p1) << 16);
          *reinterpret_cast<unsigned*>(&sP[i * LDP + j]) = pk;
        }
      } else {
#pragma unroll
        for (int jj = 0; jj < 16; jj += 2)
          *reinterpret_cast<unsigned*>(&sP[i * LDP + j0 + jj]) = 0u;
      }
    }
    __syncthreads();

    // O_head = P * V : 8 tiles, 1 per wave, K=64 -> 2 WMMAs (B from V^T rows)
    {
      int prt = wave >> 1, pct = wave & 1;
      floatx8 o = fzero();
#pragma unroll
      for (int ks = 0; ks < 2; ++ks) {
        bf16x16 ap = frag_rowK<LDP>(sP, prt * 16, ks * 32, lane);
        bf16x16 bv = frag_rowK<LDV>(sVt, pct * 16, ks * 32, lane);
        o = wmma_bf16(ap, bv, o);
      }
      const int col = pct * 16 + (lane & 15);
      const int rbase = prt * 16 + ((lane >> 4) << 3);
#pragma unroll
      for (int r = 0; r < 8; ++r)
        sO[(rbase + r) * LDA + h * HDIM + col] = f2bf_u(o[r]);
    }
    __syncthreads();
  }  // heads

  // ---- output projection: out = sO(49x384) @ proj_w + proj_b ----
  float* ob = out + (size_t)b * WIN_N * CDIM;
  for (int chunk = 0; chunk < 4; ++chunk) {   // 4 x 96 output columns
    auto stage_proj_panel = [&](int buf, int ks) {
      unsigned short* dst = sWt + buf * WPANEL;
      for (int idx = tid; idx < 96 * 4; idx += 256) {
        int c = idx >> 2, q = idx & 3;
        int gcol = chunk * 96 + c;
        const unsigned short* src = wpT + (size_t)gcol * CDIM + ks * 32 + q * 8;
        unsigned lds_off = (unsigned)(uintptr_t)(&dst[c * LDK + q * 8]);
        async_copy_b128(lds_off, src);
      }
    };

    if (chunk + 1 < 4) {   // warm GL2 for next chunk
      const unsigned short* nxt =
          wpT + (size_t)((chunk + 1) * 96 + (tid % 96)) * CDIM;
      __builtin_prefetch((const void*)nxt, 0, 0);
    }

    const int rt = wave & 3;
    floatx8 acc[3];
#pragma unroll
    for (int m = 0; m < 3; ++m) acc[m] = fzero();

    stage_proj_panel(0, 0);
    async_wait0();
    __syncthreads();

    for (int ks = 0; ks < CDIM / 32; ++ks) {
      const int cur = ks & 1;
      if (ks + 1 < CDIM / 32) stage_proj_panel(cur ^ 1, ks + 1);

      const unsigned short* wbuf = sWt + cur * WPANEL;
      bf16x16 a = frag_rowK<LDA>(sO, rt * 16, ks * 32, lane);
#pragma unroll
      for (int j = 0; j < 3; ++j) {
        int ct = (wave >> 2) * 3 + j;
        bf16x16 bw = frag_rowK<LDK>(wbuf, ct * 16, 0, lane);
        acc[j] = wmma_bf16(a, bw, acc[j]);
      }
      async_wait0();
      __syncthreads();
    }

#pragma unroll
    for (int j = 0; j < 3; ++j) {
      int ct = (wave >> 2) * 3 + j;
      int col = chunk * 96 + ct * 16 + (lane & 15);
      float pb = proj_b[col];
      const int rbase = rt * 16 + ((lane >> 4) << 3);
#pragma unroll
      for (int r = 0; r < 8; ++r) {
        int row = rbase + r;
        if (row < WIN_N) ob[(size_t)row * CDIM + col] = acc[j][r] + pb;
      }
    }
  }
}

// ---------------- launch ----------------
extern "C" void kernel_launch(void* const* d_in, const int* in_sizes, int n_in,
                              void* d_out, int out_size, void* d_ws, size_t ws_size,
                              hipStream_t stream) {
  const float* x      = (const float*)d_in[0];
  const float* mask   = (const float*)d_in[1];
  const float* qkv_w  = (const float*)d_in[2];
  const float* qkv_b  = (const float*)d_in[3];
  const float* proj_w = (const float*)d_in[4];
  const float* proj_b = (const float*)d_in[5];
  const float* rpb    = (const float*)d_in[6];
  const int*   rel    = (const int*)d_in[7];
  float* out = (float*)d_out;

  const int B_ = in_sizes[0] / (WIN_N * CDIM);                // 4096
  const int nW = in_sizes[1] / (WIN_N * WIN_N);               // 1024

  // workspace: qkv_w^T bf16 | proj_w^T bf16 | gathered bias f32  (~1.24 MB)
  unsigned short* wqT = (unsigned short*)d_ws;
  unsigned short* wpT = (unsigned short*)((char*)d_ws + (size_t)CDIM * QKV_COLS * 2);
  float* biasbuf = (float*)((char*)d_ws + (size_t)CDIM * QKV_COLS * 2 +
                            (size_t)CDIM * CDIM * 2);

  const int convN = CDIM * QKV_COLS;                          // 442368
  convert_transpose_kernel<<<(convN + 255) / 256, 256, 0, stream>>>(qkv_w, proj_w,
                                                                    wqT, wpT);
  const int biasN = NHEADS * WIN_N * WIN_N;                   // 28812
  gather_bias_kernel<<<(biasN + 255) / 256, 256, 0, stream>>>(rpb, rel, biasbuf);

  fused_window_attn_kernel<<<B_, 256, 0, stream>>>(x, mask, wqT, qkv_b, wpT,
                                                   proj_b, biasbuf, out, nW);
}
